// MessagePassingLayer_22282290331997
// MI455X (gfx1250) — matrix operand order repaired and checked by reading, not compiled, not run on hardware
//
#include <hip/hip_runtime.h>
#include <math.h>

// ---------------------------------------------------------------------------
// MessagePassingLayer for MI455X (gfx1250), bf16 WMMA with f32 accumulate.
//   B=4, N=4096, H=64
//   kernel 1: m = relu(relu(h@W1+b1)@W2+b2)  -> bf16, TRANSPOSED [B][H][N] in d_ws
//   kernel 2: msg = relu(A@m); fused GRU     -> f32 in d_out
// Kernel 2 pipelines the m-tile fetch with the Tensor Data Mover (double
// buffered, TENSORcnt-synchronized) when the TDM builtin is available.
// ---------------------------------------------------------------------------

typedef __attribute__((ext_vector_type(16))) __bf16 v16bf;
typedef __attribute__((ext_vector_type(8)))  __bf16 v8bf;
typedef __attribute__((ext_vector_type(8)))  float  v8f;
typedef __attribute__((ext_vector_type(4)))  float  v4f;
typedef __attribute__((ext_vector_type(4)))  unsigned int v4u;
typedef __attribute__((ext_vector_type(4)))  int    v4i;
typedef __attribute__((ext_vector_type(8)))  int    v8i;

static constexpr int kB = 4;
static constexpr int kN = 4096;
static constexpr int kH = 64;

#if defined(__AMDGCN__) && __has_builtin(__builtin_amdgcn_tensor_load_to_lds)
#define USE_TDM 1
#warning "CDNA5 device pass: TDM tensor_load_to_lds path ENABLED"
#else
#define USE_TDM 0
#warning "CDNA5: tensor_load_to_lds builtin not visible in this pass (fallback path)"
#endif

union BF16x16 { v16bf v; v8bf h[2]; };

__device__ __forceinline__ __bf16 to_bf16(float x) { return (__bf16)x; }

__device__ __forceinline__ v8f vzero8() {
  v8f z;
#pragma unroll
  for (int i = 0; i < 8; ++i) z[i] = 0.0f;
  return z;
}

// A-operand (16x32 bf16): lane L -> row L%16, K = {lo..lo+7, lo+16..lo+23}
__device__ __forceinline__ v16bf load_a_op(const __bf16* rowptr, int lo) {
  BF16x16 a;
  a.h[0] = *(const v8bf*)(rowptr + lo);
  a.h[1] = *(const v8bf*)(rowptr + lo + 16);
  return a.v;
}

// B-operand (32x16 bf16): lane L -> col L%16, K = {kb..kb+7, kb+16..kb+23}
__device__ __forceinline__ v16bf load_b_op(const __bf16* colptr, int kb) {
  BF16x16 b;
  b.h[0] = *(const v8bf*)(colptr + kb);
  b.h[1] = *(const v8bf*)(colptr + kb + 16);
  return b.v;
}

__device__ __forceinline__ v8f wmma_bf16(v16bf a, v16bf b, v8f c) {
  return __builtin_amdgcn_wmma_f32_16x16x32_bf16(false, a, false, b,
                                                 (short)0, c, false, false);
}

__device__ __forceinline__ float sigmoidf_(float x) {
  return 1.0f / (1.0f + __expf(-x));
}

// ---------------------------------------------------------------------------
// Kernel 1: 2-layer MLP.  One block = 64 rows of the flattened [B*N, H] input.
// Output written TRANSPOSED: mT[b][c][n]  (so kernel 2 tiles are rectangular).
// ---------------------------------------------------------------------------
__global__ __launch_bounds__(128)
void mlp_kernel(const float* __restrict__ hin,
                const float* __restrict__ W1, const float* __restrict__ b1,
                const float* __restrict__ W2, const float* __restrict__ b2,
                __bf16* __restrict__ mT) {
  __shared__ __bf16 lds_h  [64][64];   // h rows, bf16, row-major (A-op source)
  __shared__ __bf16 lds_w1t[64][64];   // lds_w1t[n][k] = W1[k][n]
  __shared__ __bf16 lds_w2t[64][64];   // lds_w2t[n][k] = W2[k][n]
  __shared__ __bf16 lds_m1 [64][64];   // layer-1 activations

  const int tid     = threadIdx.x;
  const int rowbase = blockIdx.x * 64;          // flat row in [0, B*N)

  // stage h tile (64x64 f32 -> bf16)
  {
    const int r  = tid >> 1;
    const int c0 = (tid & 1) * 32;
    const float* src = hin + (size_t)(rowbase + r) * kH + c0;
#pragma unroll
    for (int i = 0; i < 32; i += 4) {
      v4f q = *(const v4f*)(src + i);
#pragma unroll
      for (int j = 0; j < 4; ++j) lds_h[r][c0 + i + j] = to_bf16(q[j]);
    }
  }
  // stage W1^T, W2^T
  {
    const int n  = tid >> 1;
    const int k0 = (tid & 1) * 32;
#pragma unroll
    for (int i = 0; i < 32; ++i) {
      lds_w1t[n][k0 + i] = to_bf16(W1[(size_t)(k0 + i) * kH + n]);
      lds_w2t[n][k0 + i] = to_bf16(W2[(size_t)(k0 + i) * kH + n]);
    }
  }
  __syncthreads();

  const int wave = tid >> 5;
  const int lane = tid & 31;
  const int l16  = lane & 15;
  const int hi   = lane >> 4;      // 0 or 1

  // ---- layer 1 ----
  v8f acc1[4];
#pragma unroll
  for (int t = 0; t < 4; ++t) acc1[t] = vzero8();
#pragma unroll
  for (int kk = 0; kk < kH; kk += 32) {
    v16bf a = load_a_op(&lds_h[wave * 16 + l16][0], kk + hi * 8);
#pragma unroll
    for (int t = 0; t < 4; ++t) {
      v16bf b = load_b_op(&lds_w1t[t * 16 + l16][0], kk + hi * 8);
      acc1[t] = wmma_bf16(a, b, acc1[t]);
    }
  }
  // bias + relu -> LDS (re-layout C/D -> A-operand source)
#pragma unroll
  for (int t = 0; t < 4; ++t) {
    const int   col = t * 16 + l16;
    const float bv  = b1[col];
#pragma unroll
    for (int j = 0; j < 8; ++j) {
      const int mrow = wave * 16 + j + 8 * hi;
      float x = acc1[t][j] + bv;
      lds_m1[mrow][col] = to_bf16(x > 0.0f ? x : 0.0f);
    }
  }
  __syncthreads();

  // ---- layer 2 ----
  v8f acc2[4];
#pragma unroll
  for (int t = 0; t < 4; ++t) acc2[t] = vzero8();
#pragma unroll
  for (int kk = 0; kk < kH; kk += 32) {
    v16bf a = load_a_op(&lds_m1[wave * 16 + l16][0], kk + hi * 8);
#pragma unroll
    for (int t = 0; t < 4; ++t) {
      v16bf b = load_b_op(&lds_w2t[t * 16 + l16][0], kk + hi * 8);
      acc2[t] = wmma_bf16(a, b, acc2[t]);
    }
  }
  // bias + relu -> global bf16, transposed: mT[(b*kH + col)*kN + n]
#pragma unroll
  for (int t = 0; t < 4; ++t) {
    const int   col = t * 16 + l16;
    const float bv  = b2[col];
#pragma unroll
    for (int j = 0; j < 8; ++j) {
      const int grow = rowbase + wave * 16 + j + 8 * hi;   // flat row in B*N
      const int b    = grow >> 12;                          // / 4096
      const int n    = grow & (kN - 1);
      float x = acc2[t][j] + bv;
      mT[((size_t)b * kH + col) * kN + n] = to_bf16(x > 0.0f ? x : 0.0f);
    }
  }
}

// ---------------------------------------------------------------------------
// Kernel 2: msg = relu(A@m) fused with the GRU update.
// grid = (N/64, B); block = 128 threads (4 waves).
// A is streamed from HBM exactly once.  Double-buffered software pipeline:
// while tile i is consumed by WMMA, the TDM is DMA-ing tile i+1 into LDS.
// ---------------------------------------------------------------------------
__global__ __launch_bounds__(128)
void msg_gru_kernel(const float* __restrict__ A,
                    const float* __restrict__ hst,
                    const __bf16* __restrict__ mT,
                    const float* __restrict__ W_ih, const float* __restrict__ b_ih,
                    const float* __restrict__ W_hh, const float* __restrict__ b_hh,
                    float* __restrict__ out) {
  __shared__ __bf16 lds_A  [2][64][32]; // A tiles, row-major (A-op source)
  __shared__ __bf16 lds_mT [2][64][32]; // lds_mT[s][col][k] = m[k0+k][col]
  __shared__ __bf16 lds_w  [192][64];   // W_ih then W_hh (reused)
  __shared__ __bf16 lds_msg[64][64];    // relu(A@m) tile, bf16
  __shared__ __bf16 lds_h  [64][64];    // h rows, bf16

  const int tid     = threadIdx.x;
  const int rowbase = blockIdx.x * 64;        // row in [0, N)
  const int bidx    = blockIdx.y;             // batch

  const int wave = tid >> 5;
  const int lane = tid & 31;
  const int l16  = lane & 15;
  const int hi   = lane >> 4;

  // stage h state tile (bf16) — consumed only after later barriers
  {
    const int r  = tid >> 1;
    const int c0 = (tid & 1) * 32;
    const float* src = hst + ((size_t)bidx * kN + rowbase + r) * kH + c0;
#pragma unroll
    for (int i = 0; i < 32; i += 4) {
      v4f q = *(const v4f*)(src + i);
#pragma unroll
      for (int j = 0; j < 4; ++j) lds_h[r][c0 + i + j] = to_bf16(q[j]);
    }
  }

  const int   ar   = tid >> 1;                 // A-tile row this thread stages
  const int   ac0  = (tid & 1) * 16;           // 16 consecutive k's
  const float* Arow = A + ((size_t)bidx * kN + rowbase + ar) * kN + ac0;

  // stage a 64x32 A tile: f32 streamed nontemporally -> packed bf16 -> b128 LDS
  auto stage_A = [&](int buf, int k0) {
    v8bf pk0, pk1;
#pragma unroll
    for (int i = 0; i < 4; ++i) {
      v4f q = __builtin_nontemporal_load((const v4f*)(Arow + k0) + i);
#pragma unroll
      for (int j = 0; j < 4; ++j) {
        const int e = i * 4 + j;
        if (e < 8) pk0[e] = to_bf16(q[j]); else pk1[e - 8] = to_bf16(q[j]);
      }
    }
    *(v8bf*)&lds_A[buf][ar][ac0]     = pk0;
    *(v8bf*)&lds_A[buf][ar][ac0 + 8] = pk1;
    if (k0 + 32 < kN) __builtin_prefetch(Arow + k0 + 32, 0, 0);
  };

  // fetch a 64(col) x 32(k) m tile into lds_mT[buf]
  auto issue_m = [&](int buf, int k0) {
#if USE_TDM
    if (wave == 0) {
      const unsigned long long ga =
          (unsigned long long)(uintptr_t)(mT + ((size_t)bidx * kH) * kN + k0);
      const unsigned int lds_off =
          (unsigned int)(uintptr_t)&lds_mT[buf][0][0];
      v4u g0;
      g0[0] = 1u;                                    // count=1, user descriptor
      g0[1] = lds_off;                               // LDS destination (bytes)
      g0[2] = (unsigned int)(ga & 0xFFFFFFFFu);      // global_addr[31:0]
      g0[3] = (unsigned int)((ga >> 32) & 0x01FFFFFFu) | (2u << 30); // [56:32]|type=2
      v8i g1;
      g1[0] = (1 << 16);                 // workgroup_mask=0, data_size=1 (2B)
      g1[1] = (kN & 0xFFFF) << 16;       // tensor_dim0[15:0]   (bits 63:48)
      g1[2] = ((kN >> 16) & 0xFFFF) | (kH << 16); // dim0[31:16] | tensor_dim1[15:0]
      g1[3] = (32 << 16);                // tensor_dim1[31:16]=0 | tile_dim0=32
      g1[4] = kH;                        // tile_dim1=64, tile_dim2=0
      g1[5] = kN;                        // tensor_dim0_stride[31:0]
      g1[6] = 0;                         // stride hi | dim1_stride lo
      g1[7] = 0;
      v4i z4 = {0, 0, 0, 0};
#if __clang_major__ >= 23
      v8i z8 = {0, 0, 0, 0, 0, 0, 0, 0};
      __builtin_amdgcn_tensor_load_to_lds(g0, g1, z4, z4, z8, 0);
#else
      __builtin_amdgcn_tensor_load_to_lds(g0, g1, z4, z4, 0);
#endif
    }
#else
    {
      const int c   = tid >> 1;            // 0..63 (output column)
      const int kk0 = (tid & 1) * 16;      // 16 consecutive k's
      const v8bf* src =
          (const v8bf*)(mT + ((size_t)bidx * kH + c) * kN + k0 + kk0);
      *(v8bf*)&lds_mT[buf][c][kk0]     = src[0];
      *(v8bf*)&lds_mT[buf][c][kk0 + 8] = src[1];
    }
#endif
  };

  // ---- software-pipelined main loop: msg_tile += A_tile @ m_tile ----
  v8f acc[4];
#pragma unroll
  for (int t = 0; t < 4; ++t) acc[t] = vzero8();

  stage_A(0, 0);
  issue_m(0, 0);

  constexpr int kIters = kN / 32;
  for (int i = 0; i < kIters; ++i) {
    const int cur = i & 1;
    const int nxt = cur ^ 1;
    const int k0  = i * 32;

#if USE_TDM
    if (wave == 0) __builtin_amdgcn_s_wait_tensorcnt(0);  // tile cur DMA done
#endif
    __syncthreads();  // tile cur (A stores + m DMA) visible; buffer nxt retired

    if (k0 + 32 < kN) {       // kick off tile i+1 while consuming tile i
      stage_A(nxt, k0 + 32);
      issue_m(nxt, k0 + 32);
    }

    v16bf a = load_a_op(&lds_A[cur][wave * 16 + l16][0], hi * 8);
#pragma unroll
    for (int t = 0; t < 4; ++t) {
      v16bf b = load_b_op(&lds_mT[cur][t * 16 + l16][0], hi * 8);
      acc[t] = wmma_bf16(a, b, acc[t]);
    }
  }

  // relu -> lds_msg (C/D layout -> A-operand source layout)
#pragma unroll
  for (int t = 0; t < 4; ++t) {
    const int col = t * 16 + l16;
#pragma unroll
    for (int j = 0; j < 8; ++j) {
      const int mrow = wave * 16 + j + 8 * hi;
      float x = acc[t][j];
      lds_msg[mrow][col] = to_bf16(x > 0.0f ? x : 0.0f);
    }
  }

  // stage W_ih (3H x H, row-major: K already contiguous per output col)
  {
    __bf16* dst = &lds_w[0][0];
#pragma unroll
    for (int i = 0; i < 96; i += 4) {
      const int idx = tid * 96 + i;
      v4f q = *(const v4f*)(W_ih + idx);
#pragma unroll
      for (int j = 0; j < 4; ++j) dst[idx + j] = to_bf16(q[j]);
    }
  }
  __syncthreads();

  // gi = msg @ W_ih^T  (12 column tiles of 16 over 3H=192)
  v8f gi[12];
#pragma unroll
  for (int t = 0; t < 12; ++t) gi[t] = vzero8();
#pragma unroll
  for (int kk = 0; kk < kH; kk += 32) {
    v16bf a = load_a_op(&lds_msg[wave * 16 + l16][0], kk + hi * 8);
#pragma unroll
    for (int t = 0; t < 12; ++t) {
      v16bf b = load_b_op(&lds_w[t * 16 + l16][0], kk + hi * 8);
      gi[t] = wmma_bf16(a, b, gi[t]);
    }
  }
  __syncthreads();

  // swap weights: W_hh
  {
    __bf16* dst = &lds_w[0][0];
#pragma unroll
    for (int i = 0; i < 96; i += 4) {
      const int idx = tid * 96 + i;
      v4f q = *(const v4f*)(W_hh + idx);
#pragma unroll
      for (int j = 0; j < 4; ++j) dst[idx + j] = to_bf16(q[j]);
    }
  }
  __syncthreads();

  // gh = h @ W_hh^T
  v8f gh[12];
#pragma unroll
  for (int t = 0; t < 12; ++t) gh[t] = vzero8();
#pragma unroll
  for (int kk = 0; kk < kH; kk += 32) {
    v16bf a = load_a_op(&lds_h[wave * 16 + l16][0], kk + hi * 8);
#pragma unroll
    for (int t = 0; t < 12; ++t) {
      v16bf b = load_b_op(&lds_w[t * 16 + l16][0], kk + hi * 8);
      gh[t] = wmma_bf16(a, b, gh[t]);
    }
  }

  // ---- GRU gates + output (f32) ----
#pragma unroll
  for (int t = 0; t < 4; ++t) {
    const int col = t * 16 + l16;
    const float bir = b_ih[col],       bhr = b_hh[col];
    const float biz = b_ih[col + 64],  bhz = b_hh[col + 64];
    const float bin = b_ih[col + 128], bhn = b_hh[col + 128];
#pragma unroll
    for (int j = 0; j < 8; ++j) {
      const int    gr  = rowbase + wave * 16 + j + 8 * hi;
      const size_t off = ((size_t)bidx * kN + gr) * kH + col;
      const float r  = sigmoidf_((gi[t][j]     + bir) + (gh[t][j]     + bhr));
      const float z  = sigmoidf_((gi[t + 4][j] + biz) + (gh[t + 4][j] + bhz));
      const float nn = tanhf((gi[t + 8][j] + bin) + r * (gh[t + 8][j] + bhn));
      const float hv = hst[off];
      out[off] = (1.0f - z) * nn + z * hv;
    }
  }
}

// ---------------------------------------------------------------------------
extern "C" void kernel_launch(void* const* d_in, const int* in_sizes, int n_in,
                              void* d_out, int out_size, void* d_ws, size_t ws_size,
                              hipStream_t stream) {
  (void)in_sizes; (void)n_in; (void)out_size; (void)ws_size;
  const float* h   = (const float*)d_in[0];
  const float* A   = (const float*)d_in[1];
  const float* W1  = (const float*)d_in[2];
  const float* b1  = (const float*)d_in[3];
  const float* W2  = (const float*)d_in[4];
  const float* b2  = (const float*)d_in[5];
  const float* Wih = (const float*)d_in[6];
  const float* bih = (const float*)d_in[7];
  const float* Whh = (const float*)d_in[8];
  const float* bhh = (const float*)d_in[9];
  float*  out  = (float*)d_out;
  __bf16* m_ws = (__bf16*)d_ws;   // mT: B*H*N bf16 = 2 MB of scratch

  dim3 blk(128);
  mlp_kernel<<<dim3((kB * kN) / 64), blk, 0, stream>>>(h, W1, b1, W2, b2, m_ws);
  msg_gru_kernel<<<dim3(kN / 64, kB), blk, 0, stream>>>(A, h, m_ws,
                                                        Wih, bih, Whh, bhh, out);
}
